// SimplifiedTransformerBlock_69011534512524
// MI455X (gfx1250) — matrix operand order repaired and checked by reading, not compiled, not run on hardware
//
#include <hip/hip_runtime.h>
#include <hip/hip_bf16.h>
#include <math.h>

// ---- problem dims (fixed by reference) ----
#define LTOK   2304   // H*W tokens
#define DMODEL 256
#define NHEAD  8
#define HDIM   32     // DMODEL / NHEAD
#define FFDIM  1024
#define HH     48
#define WW     48
#define GN_GROUPS 32

typedef __attribute__((ext_vector_type(16))) __bf16 v16bf;
typedef __attribute__((ext_vector_type(8)))  float  v8f;
typedef __attribute__((ext_vector_type(4)))  unsigned int ui32x4;
typedef __attribute__((ext_vector_type(8)))  int          i32x8;
typedef __attribute__((ext_vector_type(4)))  int          i32x4;

#if defined(__has_builtin)
#  if __has_builtin(__builtin_amdgcn_tensor_load_to_lds) && \
      __has_builtin(__builtin_amdgcn_s_wait_tensorcnt)
#    define USE_TDM 1
#  else
#    define USE_TDM 0
#  endif
#else
#  define USE_TDM 0
#endif

#if USE_TDM
// Issue a TDM 2-D tile load: 32 (dim0, K-cols) x 64 (dim1, weight rows) f32
// elements from global (row stride = K elements) into contiguous LDS.
// 6-arg builtin form: (g0, g1, g2, g3, g4, cpol) -- groups 2..4 zero (2-D tile).
__device__ __forceinline__ void tdm_issue_w(const float* gptr, unsigned int ldsoff, int K)
{
    const unsigned long long ga = (unsigned long long)(size_t)gptr;
    ui32x4 g0;
    g0[0] = 1u;                                   // count=1, user mode
    g0[1] = ldsoff;                               // lds_addr (bytes)
    g0[2] = (unsigned int)ga;                     // global_addr[31:0]
    g0[3] = (unsigned int)((ga >> 32) & 0x01FFFFFFu) | 0x80000000u; // addr[56:32] | type=2

    const unsigned int dim0 = (unsigned int)K;    // tensor_dim0 (elements)
    const unsigned int dim1 = 64u;                // tensor_dim1 (rows)
    const unsigned long long st0 = (unsigned long long)(unsigned int)K; // dim0 stride
    i32x8 g1;
    g1[0] = (int)(2u << 16);                                    // data_size = 4B
    g1[1] = (int)((dim0 & 0xFFFFu) << 16);                      // tensor_dim0[15:0]
    g1[2] = (int)((dim0 >> 16) | ((dim1 & 0xFFFFu) << 16));     // dim0[31:16] | dim1[15:0]
    g1[3] = (int)((dim1 >> 16) | (32u << 16));                  // dim1[31:16] | tile_dim0=32
    g1[4] = (int)64u;                                           // tile_dim1=64, tile_dim2=0
    g1[5] = (int)(unsigned int)st0;                             // stride[31:0]
    g1[6] = (int)(unsigned int)((st0 >> 32) & 0xFFFFu);         // stride[47:32]
    g1[7] = 0;
    const i32x4 z4 = {0, 0, 0, 0};
    const i32x8 z8 = {0, 0, 0, 0, 0, 0, 0, 0};
    __builtin_amdgcn_tensor_load_to_lds(g0, g1, z4, z4, z8, 0);
}
#endif

// ============================================================================
// GEMM: C[M,N] = A[M,K] (f32 row-major) x W[N,K]^T (torch Linear) + bias
//       (+ optional residual).  bf16 WMMA, f32 accumulate.
// One 16x64 output tile per wave (4 WMMA per 32-K step, shared A fragment);
// 4 waves per block.  Weight panels staged LDS-side by the Tensor Data Mover
// (double-buffered, overlapped with compute) when available.
// Requires: M%16==0, N%64==0, K%32==0.
// ============================================================================
__global__ __launch_bounds__(128) void gemm_bf16_wmma(
    const float* __restrict__ A, const float* __restrict__ W,
    const float* __restrict__ bias, const float* __restrict__ res,
    float* __restrict__ C, int M, int N, int K)
{
#if USE_TDM
    __shared__ float wstage[2][4][64 * 32];       // 2 bufs x 4 waves x 8KB
#endif
    const int lane = threadIdx.x & 31;
    const int wave = threadIdx.x >> 5;
    const int ng   = N >> 6;                      // N/64 groups
    const int grp  = blockIdx.x * 4 + wave;
    if (grp >= (M >> 4) * ng) return;             // whole-wave exit
    const int mt = grp / ng;
    const int n0 = (grp % ng) << 6;
    const int hf = lane >> 4, ln = lane & 15;

    const float* arow  = A + (mt * 16 + ln) * K;  // A-frag row m = ln
    const float* wbase = W + n0 * K;              // 64 weight rows for this wave

#if USE_TDM
    const unsigned int lds0 = (unsigned int)(size_t)&wstage[0][wave][0];
    const unsigned int lds1 = (unsigned int)(size_t)&wstage[1][wave][0];
    asm volatile("" ::: "memory");
    tdm_issue_w(wbase, lds0, K);                  // prologue: stage K-step 0
    asm volatile("" ::: "memory");
#endif

    v8f acc[4] = {};
    for (int kb = 0; kb < K; kb += 32) {
        // A fragment (16x32): elem j -> K = kb+16*hf+j
        __builtin_prefetch(arow + kb + 64, 0, 0);
        v16bf af;
        {
            const float* ap = arow + kb + hf * 16;
            #pragma unroll
            for (int j = 0; j < 8; ++j) {
                af[j]     = (__bf16)ap[j];
                af[j + 8] = (__bf16)ap[8 + j];
            }
        }
#if USE_TDM
        const int buf = (kb >> 5) & 1;
        if (kb + 32 < K) {                        // stage next panel, wait current
            tdm_issue_w(wbase + kb + 32, buf ? lds0 : lds1, K);
            asm volatile("" ::: "memory");
            __builtin_amdgcn_s_wait_tensorcnt(1);
        } else {
            __builtin_amdgcn_s_wait_tensorcnt(0);
        }
        asm volatile("" ::: "memory");
        const float* wt = &wstage[buf][wave][0];  // [row 0..63][col 0..31]
        #pragma unroll
        for (int t = 0; t < 4; ++t) {
            v16bf bfv;                            // B-frag: elem j -> K=16*hf+j, col n=ln
            const float* wp = wt + (t * 16 + ln) * 32 + hf * 16;
            #pragma unroll
            for (int j = 0; j < 16; ++j) bfv[j] = (__bf16)wp[j];
            acc[t] = __builtin_amdgcn_wmma_f32_16x16x32_bf16(
                         false, af, false, bfv, (short)0, acc[t], false, false);
        }
#else
        #pragma unroll
        for (int t = 0; t < 4; ++t) {
            v16bf bfv;
            const float* wp = wbase + (t * 16 + ln) * K + kb + hf * 16;
            #pragma unroll
            for (int j = 0; j < 16; ++j) bfv[j] = (__bf16)wp[j];
            acc[t] = __builtin_amdgcn_wmma_f32_16x16x32_bf16(
                         false, af, false, bfv, (short)0, acc[t], false, false);
        }
#endif
    }

    // epilogue: elem r -> row mt*16 + 8*hf + r, cols n0 + 16*t + ln (32-bit math)
    const int base = (mt * 16 + hf * 8) * N + n0 + ln;
    if (res != nullptr) {
        #pragma unroll
        for (int t = 0; t < 4; ++t) {
            const float bv = bias[n0 + t * 16 + ln];
            #pragma unroll
            for (int r = 0; r < 8; ++r) {
                const int idx = base + r * N + t * 16;
                C[idx] = acc[t][r] + bv + res[idx];
            }
        }
    } else {
        #pragma unroll
        for (int t = 0; t < 4; ++t) {
            const float bv = bias[n0 + t * 16 + ln];
            #pragma unroll
            for (int r = 0; r < 8; ++r) {
                const int idx = base + r * N + t * 16;
                C[idx] = acc[t][r] + bv;
            }
        }
    }
}

// ============================================================================
// Flash-style MHA: O = softmax(Q K^T / sqrt(hd)) V, hd = 32, per-head slices
// of [L, D] tensors. One (head, 16-row q-tile) per wave, streaming 32 keys
// per iteration: 2 WMMA (scores) + 2 WMMA (P·V).
// ============================================================================
__global__ __launch_bounds__(256) void attn_flash_wmma(
    const float* __restrict__ Q, const float* __restrict__ K,
    const float* __restrict__ V, float* __restrict__ O,
    int Lq, int Lk)
{
    __shared__ __bf16 plds[8][16 * 32];      // per-wave P tile (bf16, 1KB each)

    const int lane = threadIdx.x & 31;
    const int wave = threadIdx.x >> 5;
    const int nqt  = Lq >> 4;
    const int gw   = blockIdx.x * 8 + wave;
    if (gw >= NHEAD * nqt) return;
    const int head = gw / nqt;
    const int qt   = gw % nqt;

    const int hf = lane >> 4;
    const int ln = lane & 15;
    const int hoff = head * HDIM;
    const float qscale = rsqrtf((float)HDIM);

    // Q fragment (A layout, 16 rows x 32 hd), pre-scaled
    v16bf qf;
    {
        const float* qrow = Q + (qt * 16 + ln) * DMODEL + hoff;
        #pragma unroll
        for (int j = 0; j < 8; ++j) {
            qf[j]     = (__bf16)(qrow[hf * 16 + j] * qscale);
            qf[j + 8] = (__bf16)(qrow[hf * 16 + 8 + j] * qscale);
        }
    }

    v8f acc0 = {}, acc1 = {};
    float rmax[8], rsum[8];
    #pragma unroll
    for (int r = 0; r < 8; ++r) { rmax[r] = -1e30f; rsum[r] = 0.0f; }

    __bf16* pl = &plds[wave][0];

    for (int kb = 0; kb < Lk; kb += 32) {
        // K^T fragments (B layout, 32 hd x 16 keys): lane = key n, elem j = hd dim
        v16bf kf0, kf1;
        {
            const float* kr0 = K + (kb + ln) * DMODEL + hoff + hf * 16;
            const float* kr1 = K + (kb + 16 + ln) * DMODEL + hoff + hf * 16;
            #pragma unroll
            for (int j = 0; j < 16; ++j) {
                kf0[j] = (__bf16)kr0[j];
                kf1[j] = (__bf16)kr1[j];
            }
        }
        v8f s0 = {}, s1 = {};
        s0 = __builtin_amdgcn_wmma_f32_16x16x32_bf16(false, qf, false, kf0, (short)0, s0, false, false);
        s1 = __builtin_amdgcn_wmma_f32_16x16x32_bf16(false, qf, false, kf1, (short)0, s1, false, false);

        // online softmax: per-row (elem r) reduction across the 16 lanes of this half
        #pragma unroll
        for (int r = 0; r < 8; ++r) {
            float v = fmaxf(s0[r], s1[r]);
            #pragma unroll
            for (int off = 1; off < 16; off <<= 1)
                v = fmaxf(v, __shfl_xor(v, off, 32));
            const float nm = fmaxf(rmax[r], v);
            const float sc = __expf(rmax[r] - nm);
            rmax[r] = nm;
            rsum[r] *= sc; acc0[r] *= sc; acc1[r] *= sc;

            const float p0 = __expf(s0[r] - nm);
            const float p1 = __expf(s1[r] - nm);
            float ps = p0 + p1;
            #pragma unroll
            for (int off = 1; off < 16; off <<= 1)
                ps += __shfl_xor(ps, off, 32);
            rsum[r] += ps;

            const int m = hf * 8 + r;             // C-layout -> LDS [row][keycol]
            pl[m * 32 + ln]      = (__bf16)p0;
            pl[m * 32 + 16 + ln] = (__bf16)p1;
        }

        // reload P as A fragment (16 q-rows x 32 keys); same-wave RAW -> dscnt wait
        v16bf pf;
        #pragma unroll
        for (int j = 0; j < 8; ++j) {
            pf[j]     = pl[ln * 32 + hf * 16 + j];
            pf[j + 8] = pl[ln * 32 + hf * 16 + 8 + j];
        }

        // V fragments (B layout, 32 keys x 16 hd-cols), two output halves
        v16bf vf0, vf1;
        #pragma unroll
        for (int j = 0; j < 16; ++j) {
            const float* vr = V + (kb + hf * 16 + j) * DMODEL + hoff;
            vf0[j] = (__bf16)vr[ln];
            vf1[j] = (__bf16)vr[16 + ln];
        }
        acc0 = __builtin_amdgcn_wmma_f32_16x16x32_bf16(false, pf, false, vf0, (short)0, acc0, false, false);
        acc1 = __builtin_amdgcn_wmma_f32_16x16x32_bf16(false, pf, false, vf1, (short)0, acc1, false, false);
    }

    #pragma unroll
    for (int r = 0; r < 8; ++r) {
        const int m = qt * 16 + hf * 8 + r;
        const float inv = 1.0f / rsum[r];
        O[m * DMODEL + hoff + ln]      = acc0[r] * inv;
        O[m * DMODEL + hoff + 16 + ln] = acc1[r] * inv;
    }
}

// ============================================================================
// LayerNorm over D=256 (one block of 256 threads per row), optional add input
// ============================================================================
__global__ __launch_bounds__(256) void layernorm_kernel(
    const float* __restrict__ x, const float* __restrict__ add,
    const float* __restrict__ g, const float* __restrict__ b,
    float* __restrict__ y)
{
    const int row = blockIdx.x;
    const int t = threadIdx.x;
    __shared__ float s1[256], s2[256];

    float v = x[row * DMODEL + t];
    if (add) v += add[row * DMODEL + t];
    s1[t] = v; s2[t] = v * v;
    __syncthreads();
    for (int off = 128; off > 0; off >>= 1) {
        if (t < off) { s1[t] += s1[t + off]; s2[t] += s2[t + off]; }
        __syncthreads();
    }
    const float mean = s1[0] * (1.0f / DMODEL);
    const float var  = s2[0] * (1.0f / DMODEL) - mean * mean;
    const float rstd = rsqrtf(var + 1e-5f);
    y[row * DMODEL + t] = (v - mean) * rstd * g[t] + b[t];
}

// ============================================================================
// GroupNorm (32 groups over FF=1024 channels, spatial 2304) + exact GELU.
// Input  x: [L, FF] row-major;  Output y: [FF, L] channel-major (for conv)
// ============================================================================
__global__ __launch_bounds__(256) void groupnorm_gelu_kernel(
    const float* __restrict__ x, const float* __restrict__ g,
    const float* __restrict__ b, float* __restrict__ y)
{
    const int grp = blockIdx.x;
    const int t = threadIdx.x;
    const int c0 = grp * (FFDIM / GN_GROUPS);         // 32 channels per group
    const int nelem = LTOK * 32;                      // 73728

    float s = 0.0f, ss = 0.0f;
    for (int i = t; i < nelem; i += 256) {
        const int l = i >> 5, cc = i & 31;
        const float v = x[l * FFDIM + c0 + cc];
        s += v; ss += v * v;
    }
    __shared__ float rs[256], rss[256];
    rs[t] = s; rss[t] = ss;
    __syncthreads();
    for (int off = 128; off > 0; off >>= 1) {
        if (t < off) { rs[t] += rs[t + off]; rss[t] += rss[t + off]; }
        __syncthreads();
    }
    const float mean = rs[0] / (float)nelem;
    const float var  = rss[0] / (float)nelem - mean * mean;
    const float rstd = rsqrtf(var + 1e-5f);

    for (int i = t; i < nelem; i += 256) {
        const int l = i >> 5, cc = i & 31;
        const int c = c0 + cc;
        float v = (x[l * FFDIM + c] - mean) * rstd * g[c] + b[c];
        v = 0.5f * v * (1.0f + erff(v * 0.70710678118654752f));   // exact GELU
        y[c * LTOK + l] = v;
    }
}

// ============================================================================
// Depthwise 5x5 conv, padding 2.
// Input  x: [FF, H, W];  kernel k: [FF, 1, 5, 5];  Output y: [L, FF]
// ============================================================================
__global__ __launch_bounds__(256) void dwconv5_kernel(
    const float* __restrict__ x, const float* __restrict__ k,
    float* __restrict__ y)
{
    const int idx = blockIdx.x * 256 + threadIdx.x;
    if (idx >= FFDIM * LTOK) return;
    const int c   = idx / LTOK;
    const int rem = idx % LTOK;
    const int hy = rem / WW, wx = rem % WW;

    const float* xc = x + c * LTOK;
    const float* kc = k + c * 25;
    float s = 0.0f;
    #pragma unroll
    for (int ky = 0; ky < 5; ++ky) {
        const int yy = hy + ky - 2;
        if (yy < 0 || yy >= HH) continue;
        #pragma unroll
        for (int kx = 0; kx < 5; ++kx) {
            const int xx = wx + kx - 2;
            if (xx < 0 || xx >= WW) continue;
            s += xc[yy * WW + xx] * kc[ky * 5 + kx];
        }
    }
    y[rem * FFDIM + c] = s;
}

// ============================================================================
// launcher
// ============================================================================
extern "C" void kernel_launch(void* const* d_in, const int* in_sizes, int n_in,
                              void* d_out, int out_size, void* d_ws, size_t ws_size,
                              hipStream_t stream)
{
    const float* tgt   = (const float*)d_in[0];
    const float* gK    = (const float*)d_in[1];
    const float* gV    = (const float*)d_in[2];
    const float* lK    = (const float*)d_in[3];
    const float* lV    = (const float*)d_in[4];
    const float* ln1_g = (const float*)d_in[5];  const float* ln1_b = (const float*)d_in[6];
    const float* ln2_g = (const float*)d_in[7];  const float* ln2_b = (const float*)d_in[8];
    const float* ln3_g = (const float*)d_in[9];  const float* ln3_b = (const float*)d_in[10];
    const float* ln4_g = (const float*)d_in[11]; const float* ln4_b = (const float*)d_in[12];
    const float* sa_Wq = (const float*)d_in[13]; const float* sa_bq = (const float*)d_in[14];
    const float* sa_Wk = (const float*)d_in[15]; const float* sa_bk = (const float*)d_in[16];
    const float* sa_Wv = (const float*)d_in[17]; const float* sa_bv = (const float*)d_in[18];
    const float* sa_Wo = (const float*)d_in[19]; const float* sa_bo = (const float*)d_in[20];
    const float* Wq    = (const float*)d_in[21]; const float* bq    = (const float*)d_in[22];
    const float* lt_Wo = (const float*)d_in[23]; const float* lt_bo = (const float*)d_in[24];
    const float* st_Wo = (const float*)d_in[25]; const float* st_bo = (const float*)d_in[26];
    const float* W1    = (const float*)d_in[27]; const float* b1    = (const float*)d_in[28];
    const float* gn_g  = (const float*)d_in[29]; const float* gn_b  = (const float*)d_in[30];
    const float* dw_k  = (const float*)d_in[31];
    const float* W2    = (const float*)d_in[32]; const float* b2    = (const float*)d_in[33];

    float* ws = (float*)d_ws;
    const size_t LD  = (size_t)LTOK * DMODEL;   // 589,824 floats
    const size_t LFF = (size_t)LTOK * FFDIM;    // 2,359,296 floats
    float* t0 = ws + 0 * LD;
    float* t1 = ws + 1 * LD;
    float* t2 = ws + 2 * LD;
    float* t3 = ws + 3 * LD;
    float* t4 = ws + 4 * LD;
    float* t5 = ws + 5 * LD;
    float* t6 = ws + 6 * LD;
    float* f0 = ws + 7 * LD;           // [L, FF]  linear1 out
    float* f1 = f0 + LFF;              // [FF, L]  GN+GELU out
    float* f2 = f1 + LFF;              // [L, FF]  conv out
    (void)in_sizes; (void)n_in; (void)out_size; (void)ws_size;

    auto gemm = [&](const float* A, const float* W, const float* bias,
                    const float* res, float* C, int M, int N, int K) {
        const int groups = (M / 16) * (N / 64);
        const int blocks = (groups + 3) / 4;
        gemm_bf16_wmma<<<blocks, 128, 0, stream>>>(A, W, bias, res, C, M, N, K);
    };
    auto attn = [&](const float* Qp, const float* Kp, const float* Vp,
                    float* Op, int Lk) {
        const int blocks = (NHEAD * (LTOK / 16)) / 8;   // 144
        attn_flash_wmma<<<blocks, 256, 0, stream>>>(Qp, Kp, Vp, Op, LTOK, Lk);
    };

    // ---- stage 1: self-attention ----
    layernorm_kernel<<<LTOK, 256, 0, stream>>>(tgt, nullptr, ln1_g, ln1_b, t0);
    gemm(t0, sa_Wq, sa_bq, nullptr, t1, LTOK, DMODEL, DMODEL);
    gemm(t0, sa_Wk, sa_bk, nullptr, t2, LTOK, DMODEL, DMODEL);
    gemm(t0, sa_Wv, sa_bv, nullptr, t3, LTOK, DMODEL, DMODEL);
    attn(t1, t2, t3, t4, LTOK);
    gemm(t4, sa_Wo, sa_bo, tgt, t5, LTOK, DMODEL, DMODEL);      // tgt1 = tgt + proj

    // ---- stage 2: long-term + short-term memory cross-attention ----
    layernorm_kernel<<<LTOK, 256, 0, stream>>>(t5, nullptr, ln2_g, ln2_b, t0);  // _tgt
    gemm(t0, Wq, bq, nullptr, t1, LTOK, DMODEL, DMODEL);         // curr_Q (== curr_K)
    attn(t1, gK, gV, t2, 2 * LTOK);                              // long-term, Lk = T*L
    gemm(t2, lt_Wo, lt_bo, t5, t6, LTOK, DMODEL, DMODEL);        // tgt1 + lt_proj
    layernorm_kernel<<<LTOK, 256, 0, stream>>>(lK, t1, ln4_g, ln4_b, t3);  // k_st
    layernorm_kernel<<<LTOK, 256, 0, stream>>>(lV, t0, ln4_g, ln4_b, t4);  // v_st
    attn(t1, t3, t4, t2, LTOK);                                  // short-term
    gemm(t2, st_Wo, st_bo, t6, t5, LTOK, DMODEL, DMODEL);        // tgt2 total

    // ---- stage 3: FFN with GN + GELU + depthwise conv ----
    layernorm_kernel<<<LTOK, 256, 0, stream>>>(t5, nullptr, ln3_g, ln3_b, t0);
    gemm(t0, W1, b1, nullptr, f0, LTOK, FFDIM, DMODEL);
    groupnorm_gelu_kernel<<<GN_GROUPS, 256, 0, stream>>>(f0, gn_g, gn_b, f1);
    dwconv5_kernel<<<(FFDIM * LTOK) / 256, 256, 0, stream>>>(f1, dw_k, f2);
    gemm(f2, W2, b2, t5, (float*)d_out, LTOK, DMODEL, FFDIM);
}
